// MultiAttnLayer_84456236908917
// MI455X (gfx1250) — compile-verified, matrix operand order checked by reading, not run on hardware
//
#include <hip/hip_runtime.h>

#define BB 8
#define SS 4096
#define DM 512
#define NH 8
#define HD 64
#define WW 128
#define CC (SS / WW)        // 32 chunks
#define TOK (BB * SS)       // 32768 tokens
#define SK (3 * WW)         // 384 keys per chunk window

typedef __attribute__((ext_vector_type(16))) __bf16 v16bf;
typedef __attribute__((ext_vector_type(8)))  __bf16 v8bf;
typedef __attribute__((ext_vector_type(8)))  float  v8f;
typedef __attribute__((ext_vector_type(4)))  float  v4f;

__device__ __forceinline__ v8f wmma_bf16(v16bf a, v16bf b, v8f c) {
  // D = A(16x32 bf16) * B(32x16 bf16) + C(16x16 f32)
  return __builtin_amdgcn_wmma_f32_16x16x32_bf16(false, a, false, b, (short)0, c,
                                                 false, false);
}

__device__ __forceinline__ v8f zero_v8f() {
  v8f z = {0.f, 0.f, 0.f, 0.f, 0.f, 0.f, 0.f, 0.f};
  return z;
}

// ---------------- prep kernels ----------------

// 8 elements per thread: b128 in, b128 out
__global__ void k_cvt_bf16(const float* __restrict__ x, __bf16* __restrict__ y, int n8) {
  int i = blockIdx.x * blockDim.x + threadIdx.x;
  if (i >= n8) return;
  v4f x0 = *(const v4f*)(x + (size_t)i * 8);
  v4f x1 = *(const v4f*)(x + (size_t)i * 8 + 4);
  v8bf o;
#pragma unroll
  for (int e = 0; e < 4; ++e) {
    o[e]     = (__bf16)x0[e];
    o[e + 4] = (__bf16)x1[e];
  }
  *(v8bf*)(y + (size_t)i * 8) = o;
}

// wt[n*DM + k] = w[k*DM + n] * scale   (bf16, N-major so B-fragments are contiguous)
__global__ void k_prep_w(const float* __restrict__ w, __bf16* __restrict__ wt, float scale) {
  int i = blockIdx.x * blockDim.x + threadIdx.x;
  if (i >= DM * DM) return;
  int k = i % DM, n = i / DM;
  wt[i] = (__bf16)(w[k * DM + n] * scale);
}

__global__ void k_prep_b(const float* __restrict__ b, float* __restrict__ bs, float scale) {
  int i = blockIdx.x * blockDim.x + threadIdx.x;
  if (i < DM) bs[i] = b[i] * scale;
}

// ---------------- QKV projection GEMM (M=32768, N=512, K=512) ----------------
// 256 threads = 8 waves; wave computes a 16(M) x 64(N) tile = 4 accumulators.
// Fragments are double-buffered across the K loop to overlap loads with WMMAs.

__global__ void k_gemm_proj(const __bf16* __restrict__ X, const __bf16* __restrict__ Wt,
                            const float* __restrict__ bias, __bf16* __restrict__ out,
                            int v_layout) {
  const int lane = threadIdx.x & 31;
  const int wave = threadIdx.x >> 5;            // 0..7
  const int l16  = lane & 15;
  const int half = lane >> 4;                   // selects K half of the 16x32 fragment
  const int m0 = blockIdx.x * 128 + wave * 16;
  const int n0 = blockIdx.y * 64;

  v8f acc[4] = {zero_v8f(), zero_v8f(), zero_v8f(), zero_v8f()};
  const __bf16* ap = X  + (size_t)(m0 + l16) * DM + half * 16;  // A: lane = row
  const __bf16* bp = Wt + (size_t)(n0 + l16) * DM + half * 16;  // B: lane = column

  v16bf a_cur = *(const v16bf*)ap;
  v16bf b_cur[4];
#pragma unroll
  for (int n = 0; n < 4; ++n)
    b_cur[n] = *(const v16bf*)(bp + (size_t)(n * 16) * DM);

  for (int kk = 0; kk < DM - 32; kk += 32) {
    const int kn = kk + 32;
    v16bf a_nxt = *(const v16bf*)(ap + kn);       // prefetch next K-step
    v16bf b_nxt[4];
#pragma unroll
    for (int n = 0; n < 4; ++n)
      b_nxt[n] = *(const v16bf*)(bp + (size_t)(n * 16) * DM + kn);
#pragma unroll
    for (int n = 0; n < 4; ++n)
      acc[n] = wmma_bf16(a_cur, b_cur[n], acc[n]);
    a_cur = a_nxt;
#pragma unroll
    for (int n = 0; n < 4; ++n) b_cur[n] = b_nxt[n];
  }
#pragma unroll
  for (int n = 0; n < 4; ++n)                     // last K-step
    acc[n] = wmma_bf16(a_cur, b_cur[n], acc[n]);

#pragma unroll
  for (int n = 0; n < 4; ++n) {
    const int col = n0 + n * 16 + l16;
    const float bv = bias[col];
#pragma unroll
    for (int r = 0; r < 8; ++r) {
      const int m = m0 + half * 8 + r;            // C layout: lanes>=16 hold M=8..15
      float v = acc[n][r] + bv;
      if (v_layout) {                              // V: write [b, h, dim, token]
        int bb = m / SS, t = m % SS;
        int hh = col / HD, dd = col % HD;
        out[((size_t)((bb * NH + hh) * HD) + dd) * SS + t] = (__bf16)v;
      } else {                                     // Q/K: [token, d_model]
        out[(size_t)m * DM + col] = (__bf16)v;
      }
    }
  }
}

// ---------------- sliding-window attention ----------------
// One wave per (b, h, c); loops over 8 query tiles of 16 rows.

__global__ void k_attn(const __bf16* __restrict__ Qb, const __bf16* __restrict__ Kb,
                       const __bf16* __restrict__ Vt, float* __restrict__ attn) {
  __shared__ __align__(32) float  sc[16 * SK];   // scores, f32
  __shared__ __align__(32) __bf16 sp[16 * SK];   // softmax probs, bf16 (A of PV gemm)
  __shared__ float rowsum[16];

  const int lane = threadIdx.x;                  // blockDim.x == 32
  const int l16  = lane & 15;
  const int half = lane >> 4;
  const int c = blockIdx.x % CC;
  const int h = (blockIdx.x / CC) % NH;
  const int b = blockIdx.x / (CC * NH);
  const int kwin0 = c * WW - WW;                 // first global key of the 3w window

  // loop-invariant base of this lane's K fragment column stream
  const size_t khbase = (size_t)b * SS * DM + (size_t)h * HD + half * 16;

  for (int qt = 0; qt < 8; ++qt) {
    const int qrow0 = c * WW + qt * 16;

    // ---- phase 1: scores = Q (16x64) * K^T (64x384), pipelined over 24 tiles ----
    const __bf16* qp = Qb + (size_t)(b * SS + qrow0 + l16) * DM + h * HD + half * 16;
    v16bf a0 = *(const v16bf*)qp;
    v16bf a1 = *(const v16bf*)(qp + 32);

    {
      int jg0 = kwin0 + l16;
      int jc0 = jg0 < 0 ? 0 : (jg0 >= SS ? SS - 1 : jg0);
      const __bf16* kp = Kb + khbase + (size_t)jc0 * DM;
      v16bf b0 = *(const v16bf*)kp;
      v16bf b1 = *(const v16bf*)(kp + 32);
      for (int yt = 0; yt < SK / 16 - 1; ++yt) {
        int jg = kwin0 + (yt + 1) * 16 + l16;     // prefetch next key tile
        int jc = jg < 0 ? 0 : (jg >= SS ? SS - 1 : jg);
        const __bf16* kn = Kb + khbase + (size_t)jc * DM;
        v16bf n0 = *(const v16bf*)kn;
        v16bf n1 = *(const v16bf*)(kn + 32);
        v8f s = zero_v8f();
        s = wmma_bf16(a0, b0, s);
        s = wmma_bf16(a1, b1, s);
#pragma unroll
        for (int r = 0; r < 8; ++r)
          sc[(r + half * 8) * SK + yt * 16 + l16] = s[r];
        b0 = n0; b1 = n1;
      }
      v8f s = zero_v8f();                         // last tile
      s = wmma_bf16(a0, b0, s);
      s = wmma_bf16(a1, b1, s);
#pragma unroll
      for (int r = 0; r < 8; ++r)
        sc[(r + half * 8) * SK + (SK / 16 - 1) * 16 + l16] = s[r];
    }
    __syncthreads();

    // ---- phase 2: masked softmax; lane pair (l16, l16+16) owns row l16 ----
    {
      const int ig = qrow0 + l16;
      const int y0 = half * (SK / 2);
      float mx = -3.0e38f;
      for (int y = y0; y < y0 + SK / 2; ++y) {
        int jg = kwin0 + y;
        bool ok = (jg >= 0) && (jg < SS) && (jg - ig <= WW) && (ig - jg <= WW);
        float v = ok ? sc[l16 * SK + y] : -3.0e38f;
        mx = fmaxf(mx, v);
      }
      mx = fmaxf(mx, __shfl_xor(mx, 16, 32));
      float sum = 0.f;
      for (int y = y0; y < y0 + SK / 2; ++y) {
        int jg = kwin0 + y;
        bool ok = (jg >= 0) && (jg < SS) && (jg - ig <= WW) && (ig - jg <= WW);
        float e = ok ? __expf(sc[l16 * SK + y] - mx) : 0.f;
        sp[l16 * SK + y] = (__bf16)e;
        sum += e;
      }
      sum += __shfl_xor(sum, 16, 32);
      if (half == 0) rowsum[l16] = sum;
    }
    __syncthreads();

    // ---- phase 3: out = P (16x384) * V (384x64), pipelined over 12 K-steps ----
    v8f o[4] = {zero_v8f(), zero_v8f(), zero_v8f(), zero_v8f()};
    {
      const __bf16* vhb = Vt + ((size_t)((b * NH + h) * HD) + l16) * SS;
      int kb0 = kwin0 + half * 16;
      int t00 = kb0 < 0 ? 0 : (kb0 > SS - 16 ? SS - 16 : kb0);
      v16bf pa = *(const v16bf*)(sp + l16 * SK + half * 16);
      v16bf vf[4];
#pragma unroll
      for (int n = 0; n < 4; ++n)
        vf[n] = *(const v16bf*)(vhb + (size_t)(n * 16) * SS + t00);

      for (int kt = 0; kt < SK / 32 - 1; ++kt) {
        int kb = kwin0 + (kt + 1) * 32 + half * 16;     // prefetch next K-step
        int t0 = kb < 0 ? 0 : (kb > SS - 16 ? SS - 16 : kb);
        v16bf pn = *(const v16bf*)(sp + l16 * SK + (kt + 1) * 32 + half * 16);
        v16bf vn[4];
#pragma unroll
        for (int n = 0; n < 4; ++n)
          vn[n] = *(const v16bf*)(vhb + (size_t)(n * 16) * SS + t0);
#pragma unroll
        for (int n = 0; n < 4; ++n)
          o[n] = wmma_bf16(pa, vf[n], o[n]);
        pa = pn;
#pragma unroll
        for (int n = 0; n < 4; ++n) vf[n] = vn[n];
      }
#pragma unroll
      for (int n = 0; n < 4; ++n)                        // last K-step
        o[n] = wmma_bf16(pa, vf[n], o[n]);
    }
#pragma unroll
    for (int n = 0; n < 4; ++n) {
#pragma unroll
      for (int r = 0; r < 8; ++r) {
        int m = r + half * 8;
        float val = o[n][r] / rowsum[m];
        attn[(size_t)(b * SS + qrow0 + m) * DM + h * HD + n * 16 + l16] = val;
      }
    }
    __syncthreads();
  }
}

// ---------------- residual + layernorm: one wave per token, b128 traffic ----------------

__global__ void k_ln(const float* __restrict__ x, const float* __restrict__ a,
                     const float* __restrict__ g, const float* __restrict__ be,
                     float* __restrict__ out) {
  const int lane = threadIdx.x & 31;
  const size_t tok = (size_t)blockIdx.x * 8 + (threadIdx.x >> 5);
  const float* xr = x + tok * DM;
  const float* ar = a + tok * DM;
  v4f v[4];
  float s = 0.f, s2 = 0.f;
#pragma unroll
  for (int j = 0; j < 4; ++j) {
    const int idx = j * 128 + lane * 4;
    v4f xv = *(const v4f*)(xr + idx);
    v4f av = *(const v4f*)(ar + idx);
    v4f t = xv + av;
    v[j] = t;
#pragma unroll
    for (int e = 0; e < 4; ++e) { s += t[e]; s2 += t[e] * t[e]; }
  }
#pragma unroll
  for (int off = 16; off > 0; off >>= 1) {
    s  += __shfl_xor(s,  off, 32);
    s2 += __shfl_xor(s2, off, 32);
  }
  const float mean = s * (1.f / DM);
  const float var  = s2 * (1.f / DM) - mean * mean;
  const float inv  = rsqrtf(var + 1e-5f);
#pragma unroll
  for (int j = 0; j < 4; ++j) {
    const int idx = j * 128 + lane * 4;
    v4f gw = *(const v4f*)(g + idx);
    v4f bw = *(const v4f*)(be + idx);
    v4f r;
#pragma unroll
    for (int e = 0; e < 4; ++e) r[e] = (v[j][e] - mean) * inv * gw[e] + bw[e];
    *(v4f*)(out + tok * DM + idx) = r;
  }
}

// ---------------- launch ----------------

extern "C" void kernel_launch(void* const* d_in, const int* in_sizes, int n_in,
                              void* d_out, int out_size, void* d_ws, size_t ws_size,
                              hipStream_t stream) {
  (void)in_sizes; (void)n_in; (void)out_size; (void)ws_size;
  const float* src_q = (const float*)d_in[0];
  // d_in[1] = src_kv: unused by the reference forward pass
  const float* Wq = (const float*)d_in[2]; const float* bq = (const float*)d_in[3];
  const float* Wk = (const float*)d_in[4]; const float* bk = (const float*)d_in[5];
  const float* Wv = (const float*)d_in[6]; const float* bv = (const float*)d_in[7];
  const float* ln_w = (const float*)d_in[8]; const float* ln_b = (const float*)d_in[9];
  float* out = (float*)d_out;

  char* p = (char*)d_ws;
  __bf16* Xb  = (__bf16*)p; p += (size_t)TOK * DM * 2;
  __bf16* Qb  = (__bf16*)p; p += (size_t)TOK * DM * 2;
  __bf16* Kb  = (__bf16*)p; p += (size_t)TOK * DM * 2;
  __bf16* Vt  = (__bf16*)p; p += (size_t)TOK * DM * 2;   // [b, h, dim, token]
  __bf16* Wtq = (__bf16*)p; p += (size_t)DM * DM * 2;
  __bf16* Wtk = (__bf16*)p; p += (size_t)DM * DM * 2;
  __bf16* Wtv = (__bf16*)p; p += (size_t)DM * DM * 2;
  float*  bqs = (float*)p;  p += DM * 4;
  float*  bks = (float*)p;  p += DM * 4;
  float*  bvs = (float*)p;  p += DM * 4;
  float*  attn = (float*)p; p += (size_t)TOK * DM * 4;

  const float qscale = 0.125f;                   // 1/sqrt(HD), folded into Wq/bq

  const int n8 = TOK * DM / 8;
  k_cvt_bf16<<<(n8 + 255) / 256, 256, 0, stream>>>(src_q, Xb, n8);
  k_prep_w<<<(DM * DM + 255) / 256, 256, 0, stream>>>(Wq, Wtq, qscale);
  k_prep_w<<<(DM * DM + 255) / 256, 256, 0, stream>>>(Wk, Wtk, 1.0f);
  k_prep_w<<<(DM * DM + 255) / 256, 256, 0, stream>>>(Wv, Wtv, 1.0f);
  k_prep_b<<<2, 256, 0, stream>>>(bq, bqs, qscale);
  k_prep_b<<<2, 256, 0, stream>>>(bk, bks, 1.0f);
  k_prep_b<<<2, 256, 0, stream>>>(bv, bvs, 1.0f);

  dim3 g(TOK / 128, DM / 64);
  k_gemm_proj<<<g, 256, 0, stream>>>(Xb, Wtq, bqs, Qb, 0);
  k_gemm_proj<<<g, 256, 0, stream>>>(Xb, Wtk, bks, Kb, 0);
  k_gemm_proj<<<g, 256, 0, stream>>>(Xb, Wtv, bvs, Vt, 1);

  k_attn<<<BB * NH * CC, 32, 0, stream>>>(Qb, Kb, Vt, attn);

  k_ln<<<TOK / 8, 256, 0, stream>>>(src_q, attn, ln_w, ln_b, out);
}
